// CNN_Medium_51728586113106
// MI455X (gfx1250) — 8-bit matrix operand order repaired and checked by reading, compiled, not run on hardware
//
#include <hip/hip_runtime.h>
#include <math.h>

typedef __attribute__((ext_vector_type(8))) int v8i;

#define BNEPS 1e-3f

static constexpr size_t SZ_A2   = 512ULL * 34 * 34 * 128;
static constexpr size_t SZ_A3   = 512ULL * 18 * 18 * 128;
static constexpr size_t SZ_A4   = 512ULL * 18 * 18 * 256;
static constexpr size_t SZ_A5   = 512ULL * 10 * 10 * 256;
static constexpr size_t SZ_A6   = 512ULL * 10 * 10 * 512;
static constexpr size_t SZ_RAW  = 512ULL * 32 * 32 * 128 * 2;
static constexpr size_t OFF_ACTA = 0;
static constexpr size_t OFF_ACTB = OFF_ACTA + SZ_A2;
static constexpr size_t OFF_RAW  = OFF_ACTB + SZ_A3;
static constexpr size_t OFF_WB2  = OFF_RAW + SZ_RAW;
static constexpr size_t OFF_WB3  = OFF_WB2 + 9ULL * 128 * 128;
static constexpr size_t OFF_WB4  = OFF_WB3 + 9ULL * 256 * 128;
static constexpr size_t OFF_WB5  = OFF_WB4 + 9ULL * 256 * 256;
static constexpr size_t OFF_WB6  = OFF_WB5 + 9ULL * 512 * 256;
static constexpr size_t OFF_WB7  = OFF_WB6 + 9ULL * 512 * 512;
static constexpr size_t OFF_T1   = OFF_WB7 + 16ULL * 16 * 512;
static constexpr size_t OFF_T2   = OFF_T1 + 128 * 4;
static constexpr size_t OFF_T3   = OFF_T2 + 128 * 4;
static constexpr size_t OFF_T4   = OFF_T3 + 256 * 4;
static constexpr size_t OFF_T5   = OFF_T4 + 256 * 4;
static constexpr size_t OFF_T6   = OFF_T5 + 512 * 4;
static constexpr size_t OFF_LOG  = OFF_T6 + 512 * 4;


__global__ void thresh_kernel(const float* __restrict__ g, const float* __restrict__ be,
                              const float* __restrict__ m, const float* __restrict__ v,
                              float* __restrict__ t, int C) {
  int c = blockIdx.x * blockDim.x + threadIdx.x;
  if (c < C) t[c] = m[c] - be[c] * sqrtf(v[c] + BNEPS) / g[c];
}

__global__ void packw_kernel(const float* __restrict__ w, signed char* __restrict__ wb,
                             int taps, int CIN, int COUTA, int COUTP) {
  long long i = blockIdx.x * (long long)blockDim.x + threadIdx.x;
  long long total = (long long)taps * COUTP * CIN;
  if (i >= total) return;
  int ci = (int)(i % CIN);
  long long r = i / CIN;
  int co = (int)(r % COUTP);
  int tap = (int)(r / COUTP);
  signed char s = 0;
  if (co < COUTA) {
    float wf = w[((long long)tap * CIN + ci) * COUTA + co];
    s = (wf >= 0.f) ? 1 : -1;
  }
  wb[i] = s;
}

__global__ void conv1_kernel(const float* __restrict__ x, const float* __restrict__ w1,
                             const float* __restrict__ b1, const float* __restrict__ t1,
                             signed char* __restrict__ a2) {
  long long i = blockIdx.x * (long long)blockDim.x + threadIdx.x;
  const long long total = 512LL * 34 * 34 * 128;
  if (i >= total) return;
  int co = (int)(i & 127);
  long long r = i >> 7;
  int xx = (int)(r % 34); r /= 34;
  int yy = (int)(r % 34);
  int n  = (int)(r / 34);
  signed char s = 1;
  if (yy >= 2 && yy <= 31 && xx >= 2 && xx <= 31) {
    int oy = yy - 2, ox = xx - 2;
    float acc = b1[co];
    #pragma unroll
    for (int ky = 0; ky < 3; ++ky)
      #pragma unroll
      for (int kx = 0; kx < 3; ++kx) {
        const float* xp = x + (((long long)n * 32 + (oy + ky)) * 32 + (ox + kx)) * 3;
        const float* wp = w1 + (long long)((ky * 3 + kx) * 3) * 128 + co;
        acc = fmaf(xp[0], wp[0], acc);
        acc = fmaf(xp[1], wp[128], acc);
        acc = fmaf(xp[2], wp[256], acc);
      }
    float v = fmaxf(acc, 0.f);
    s = (v >= t1[co]) ? 1 : -1;
  }
  a2[i] = s;
}

template <int PAD>
__global__ void pool_thresh_kernel(const short* __restrict__ raw, const float* __restrict__ thr,
                                   signed char* __restrict__ out, int nimg, int H, int W, int C) {
  long long i = blockIdx.x * (long long)blockDim.x + threadIdx.x;
  int HP = H / 2, WP = W / 2;
  long long total = (long long)nimg * HP * WP * C;
  if (i >= total) return;
  int c = (int)(i % C);
  long long r = i / C;
  int x2 = (int)(r % WP); r /= WP;
  int y2 = (int)(r % HP);
  int n  = (int)(r / HP);
  const short* p0 = raw + (((long long)n * H + 2 * y2) * W + 2 * x2) * C + c;
  int v = p0[0];
  int v1 = p0[C];                      if (v1 > v) v = v1;
  int v2 = p0[(long long)W * C];       if (v2 > v) v = v2;
  int v3 = p0[(long long)W * C + C];   if (v3 > v) v = v3;
  signed char s = ((float)v >= thr[c]) ? 1 : -1;
  out[(((long long)n * (HP + 2 * PAD) + y2 + PAD) * (WP + 2 * PAD) + x2 + PAD) * C + c] = s;
}

__global__ void bn_softmax_kernel(const float* __restrict__ logits,
                                  const float* __restrict__ g, const float* __restrict__ be,
                                  const float* __restrict__ m, const float* __restrict__ v,
                                  float* __restrict__ out, int nimg) {
  int n = blockIdx.x * blockDim.x + threadIdx.x;
  if (n >= nimg) return;
  float z[10];
  float mx = -1e30f;
  #pragma unroll
  for (int c = 0; c < 10; ++c) {
    float l = logits[n * 10 + c];
    float zz = g[c] * (l - m[c]) * rsqrtf(v[c] + BNEPS) + be[c];
    z[c] = zz;
    if (zz > mx) mx = zz;
  }
  float s = 0.f;
  #pragma unroll
  for (int c = 0; c < 10; ++c) { z[c] = expf(z[c] - mx); s += z[c]; }
  float inv = 1.f / s;
  #pragma unroll
  for (int c = 0; c < 10; ++c) out[n * 10 + c] = z[c] * inv;
}

static constexpr int EP_RAW16       = 0;
static constexpr int EP_THRESH_PAD1 = 1;
static constexpr int EP_LOGITS10    = 2;

template <int CIN, int COUTP, int HIN, int WIN, int HOUT, int WOUT, int KHW, int NT, int MODE>
__global__ __launch_bounds__(NT * 32)
void binconv_kernel(const signed char* __restrict__ act,
                    const signed char* __restrict__ wb,
                    const float* __restrict__ thr,
                    void* __restrict__ out, int nimg) {
  const int lane = threadIdx.x & 31;
  const int wave = threadIdx.x >> 5;
  const int row  = lane & 15;
  const int half = lane >> 4;

  const int numM = nimg * HOUT * WOUT / 16;
  const int mtile = blockIdx.x % numM;
  const int ntile = blockIdx.x / numM;
  const int co = (ntile * NT + wave) * 16 + row;

  const int p = mtile * 16 + row;
  const int n = p / (HOUT * WOUT);
  const int rem = p % (HOUT * WOUT);
  const int oy = rem / WOUT, ox = rem % WOUT;

  v8i acc = {};
  for (int ky = 0; ky < KHW; ++ky) {
    for (int kx = 0; kx < KHW; ++kx) {
      const signed char* abase =
          act + (((long long)n * HIN + (oy + ky)) * WIN + (ox + kx)) * CIN;
      const signed char* bbase =
          wb + ((long long)(ky * KHW + kx) * COUTP + co) * CIN;
      #pragma unroll
      for (int kc = 0; kc < CIN; kc += 64) {
        v8i a, b;
        #pragma unroll
        for (int v = 0; v < 8; ++v) {
          int ak = kc + ((v >> 1) * 16 + (v & 1) * 4 + half * 8);
          int bk = kc + ((v >> 1) * 16 + (v & 1) * 4 + half * 8);
          a[v] = *(const int*)(abase + ak);
          b[v] = *(const int*)(bbase + bk);
        }
        acc = __builtin_amdgcn_wmma_i32_16x16x64_iu8(true, a, true, b, acc, false, false);
      }
    }
  }

  #pragma unroll
  for (int r = 0; r < 8; ++r) {
    int M = r + half * 8;
    int pp = mtile * 16 + M;
    int nn = pp / (HOUT * WOUT);
    int rr = pp % (HOUT * WOUT);
    int yy = rr / WOUT, xx = rr % WOUT;
    int val = acc[r];
    if (val < 0) val = 0;
    if (MODE == EP_RAW16) {
      ((short*)out)[(((long long)nn * HOUT + yy) * WOUT + xx) * COUTP + co] = (short)val;
    } else if (MODE == EP_THRESH_PAD1) {
      signed char s = ((float)val >= thr[co]) ? 1 : -1;
      ((signed char*)out)[(((long long)nn * (HOUT + 2) + yy + 1) * (WOUT + 2) + xx + 1) * COUTP + co] = s;
    } else {
      if (co < 10) ((float*)out)[nn * 10 + co] = (float)val;
    }
  }
}

extern "C" void kernel_launch(void* const* d_in, const int* in_sizes, int n_in,
                              void* d_out, int out_size, void* d_ws, size_t ws_size,
                              hipStream_t stream) {
  (void)in_sizes; (void)n_in; (void)out_size; (void)ws_size;
  const float* x  = (const float*)d_in[0];
  const float* w1 = (const float*)d_in[1];
  const float* b1 = (const float*)d_in[2];
  const float* g1 = (const float*)d_in[3],  *be1 = (const float*)d_in[4],
             * m1 = (const float*)d_in[5],  *v1  = (const float*)d_in[6];
  const float* w2 = (const float*)d_in[7];
  const float* g2 = (const float*)d_in[8],  *be2 = (const float*)d_in[9],
             * m2 = (const float*)d_in[10], *v2  = (const float*)d_in[11];
  const float* w3 = (const float*)d_in[12];
  const float* g3 = (const float*)d_in[13], *be3 = (const float*)d_in[14],
             * m3 = (const float*)d_in[15], *v3  = (const float*)d_in[16];
  const float* w4 = (const float*)d_in[17];
  const float* g4 = (const float*)d_in[18], *be4 = (const float*)d_in[19],
             * m4 = (const float*)d_in[20], *v4  = (const float*)d_in[21];
  const float* w5 = (const float*)d_in[22];
  const float* g5 = (const float*)d_in[23], *be5 = (const float*)d_in[24],
             * m5 = (const float*)d_in[25], *v5  = (const float*)d_in[26];
  const float* w6 = (const float*)d_in[27];
  const float* g6 = (const float*)d_in[28], *be6 = (const float*)d_in[29],
             * m6 = (const float*)d_in[30], *v6  = (const float*)d_in[31];
  const float* w7 = (const float*)d_in[32];
  const float* g7 = (const float*)d_in[33], *be7 = (const float*)d_in[34],
             * m7 = (const float*)d_in[35], *v7  = (const float*)d_in[36];

  char* ws = (char*)d_ws;
  signed char* actA = (signed char*)(ws + OFF_ACTA);
  signed char* actB = (signed char*)(ws + OFF_ACTB);
  short*       raw  = (short*)(ws + OFF_RAW);
  signed char* wb2 = (signed char*)(ws + OFF_WB2);
  signed char* wb3 = (signed char*)(ws + OFF_WB3);
  signed char* wb4 = (signed char*)(ws + OFF_WB4);
  signed char* wb5 = (signed char*)(ws + OFF_WB5);
  signed char* wb6 = (signed char*)(ws + OFF_WB6);
  signed char* wb7 = (signed char*)(ws + OFF_WB7);
  float* t1 = (float*)(ws + OFF_T1);
  float* t2 = (float*)(ws + OFF_T2);
  float* t3 = (float*)(ws + OFF_T3);
  float* t4 = (float*)(ws + OFF_T4);
  float* t5 = (float*)(ws + OFF_T5);
  float* t6 = (float*)(ws + OFF_T6);
  float* logits = (float*)(ws + OFF_LOG);

  const int N = 512;

  auto packLaunch = [&](const float* w, signed char* wb, int taps, int cin, int couta, int coutp) {
    long long tot = (long long)taps * coutp * cin;
    packw_kernel<<<dim3((unsigned)((tot + 255) / 256)), 256, 0, stream>>>(w, wb, taps, cin, couta, coutp);
  };
  packLaunch(w2, wb2, 9, 128, 128, 128);
  packLaunch(w3, wb3, 9, 128, 256, 256);
  packLaunch(w4, wb4, 9, 256, 256, 256);
  packLaunch(w5, wb5, 9, 256, 512, 512);
  packLaunch(w6, wb6, 9, 512, 512, 512);
  packLaunch(w7, wb7, 16, 512, 10, 16);

  thresh_kernel<<<1, 128, 0, stream>>>(g1, be1, m1, v1, t1, 128);
  thresh_kernel<<<1, 128, 0, stream>>>(g2, be2, m2, v2, t2, 128);
  thresh_kernel<<<1, 256, 0, stream>>>(g3, be3, m3, v3, t3, 256);
  thresh_kernel<<<1, 256, 0, stream>>>(g4, be4, m4, v4, t4, 256);
  thresh_kernel<<<2, 256, 0, stream>>>(g5, be5, m5, v5, t5, 512);
  thresh_kernel<<<2, 256, 0, stream>>>(g6, be6, m6, v6, t6, 512);

  conv1_kernel<<<dim3(512u * 34 * 34 * 128 / 256), 256, 0, stream>>>(x, w1, b1, t1, actA);

  binconv_kernel<128, 128, 34, 34, 32, 32, 3, 4, EP_RAW16>
      <<<dim3((N * 32 * 32 / 16) * 2), 128, 0, stream>>>(actA, wb2, nullptr, raw, N);
  hipMemsetAsync(actB, 1, SZ_A3, stream);
  pool_thresh_kernel<1><<<dim3(512u * 16 * 16 * 128 / 256), 256, 0, stream>>>(raw, t2, actB, N, 32, 32, 128);

  hipMemsetAsync(actA, 1, SZ_A4, stream);
  binconv_kernel<128, 256, 18, 18, 16, 16, 3, 4, EP_THRESH_PAD1>
      <<<dim3((N * 16 * 16 / 16) * 4), 128, 0, stream>>>(actB, wb3, t3, actA, N);

  binconv_kernel<256, 256, 18, 18, 16, 16, 3, 4, EP_RAW16>
      <<<dim3((N * 16 * 16 / 16) * 4), 128, 0, stream>>>(actA, wb4, nullptr, raw, N);
  hipMemsetAsync(actB, 1, SZ_A5, stream);
  pool_thresh_kernel<1><<<dim3(512u * 8 * 8 * 256 / 256), 256, 0, stream>>>(raw, t4, actB, N, 16, 16, 256);

  hipMemsetAsync(actA, 1, SZ_A6, stream);
  binconv_kernel<256, 512, 10, 10, 8, 8, 3, 4, EP_THRESH_PAD1>
      <<<dim3((N * 8 * 8 / 16) * 8), 128, 0, stream>>>(actB, wb5, t5, actA, N);

  binconv_kernel<512, 512, 10, 10, 8, 8, 3, 4, EP_RAW16>
      <<<dim3((N * 8 * 8 / 16) * 8), 128, 0, stream>>>(actA, wb6, nullptr, raw, N);
  pool_thresh_kernel<0><<<dim3(512u * 4 * 4 * 512 / 256), 256, 0, stream>>>(raw, t6, actB, N, 8, 8, 512);

  binconv_kernel<512, 16, 4, 4, 1, 1, 4, 1, EP_LOGITS10>
      <<<dim3(N / 16), 32, 0, stream>>>(actB, wb7, nullptr, logits, N);

  bn_softmax_kernel<<<2, 256, 0, stream>>>(logits, g7, be7, m7, v7, (float*)d_out, N);
}
